// IndexForOnnx_17549236372180
// MI455X (gfx1250) — matrix operand order repaired and checked by reading, not compiled, not run on hardware
//
#include <hip/hip_runtime.h>

// ---------------------------------------------------------------------------
// fastersvc match_features (L2, k=4) for MI455X / gfx1250.
// x:     [1, 768, 2048]  fp32   (d-major, T columns)
// index: [1, 768, 16384] fp32   (d-major, N columns)
// out:   [1, 768, 2048]  fp32
//
// Pipeline:
//   1) r2_kernel:   r2[n] = ||ref_n||^2 (exact fp32)                 -> ws
//   2) match_topk:  f16 WMMA GEMM (fp32 accum), 2 accumulators/wave,
//                   fused running top-4 on key = r2 - 2*cross
//                   (s2 and sqrt dropped: monotone per-row)          -> idx ws
//   3) gather:      out[d][t] = mean of 4 selected fp32 ref columns
// ---------------------------------------------------------------------------

typedef _Float16 v2h  __attribute__((ext_vector_type(2)));
typedef _Float16 v8h  __attribute__((ext_vector_type(8)));
typedef _Float16 v16h __attribute__((ext_vector_type(16)));
typedef float    v8f  __attribute__((ext_vector_type(8)));

#define DIM    768
#define TSRC   2048
#define NREF   16384
#define KSTEPS 24                    // 768 / 32 (K per WMMA)
#define MBLK   64                    // source rows per workgroup
#define NCH    16                    // columns per WMMA
#define SCH    32                    // columns per super-chunk (2 WMMA accs)
#define NPH    2                     // n-phases (waves 0-3 / 4-7)
#define SCPP   (NREF / SCH / NPH)    // 256 super-chunks per phase

// LDS layout (in halves). A padded to stride 776 (1552B = 388 dwords, %64 = 4
// -> conflict-free b128 reads across lanes). B slab rows padded to stride 40
// (80B = 20 dwords; 16B-aligned rows, distinct banks across 16 rows).
#define A_STRIDE   776
#define A_HALVES   (MBLK * A_STRIDE)            // 49664 halves
#define B_STRIDE   40
#define B_SLABH    (SCH * B_STRIDE)             // 1280 halves per slab
#define LDS_HALVES (A_HALVES + NPH * 2 * B_SLABH)
#define LDS_BYTES  (LDS_HALVES * 2)             // 109568 B

__global__ void r2_kernel(const float* __restrict__ ref, float* __restrict__ r2) {
  int n = blockIdx.x * blockDim.x + threadIdx.x;
  float s = 0.f;
  for (int d = 0; d < DIM; ++d) { float v = ref[d * NREF + n]; s += v * v; }
  r2[n] = s;
}

__launch_bounds__(256)
__global__ void match_topk_kernel(const float* __restrict__ x,
                                  const float* __restrict__ ref,
                                  const float* __restrict__ r2,
                                  int* __restrict__ oidx) {
  extern __shared__ _Float16 lds[];
  _Float16* As = lds;
  _Float16* Bs = lds + A_HALVES;

  const int tid = threadIdx.x;
  const int wgM = blockIdx.x * MBLK;

  // ---- stage A tile (64 rows x 768 K) as f16 into LDS ----
  // Each thread handles a k-adjacent pair for one m: 1 packed ds_store_b32.
  for (int r = 0; r < (MBLK * DIM) / (256 * 2); ++r) {
    int id = r * 256 + tid;
    int k2 = id >> 6;            // pair index 0..383 -> k = 2*k2
    int m  = id & (MBLK - 1);    // 0..63, consecutive across lanes
    float a0 = x[(2 * k2 + 0) * TSRC + wgM + m];
    float a1 = x[(2 * k2 + 1) * TSRC + wgM + m];
    v2h pk = { (_Float16)a0, (_Float16)a1 };
    *(v2h*)&As[m * A_STRIDE + 2 * k2] = pk;
  }

  const int wave = tid >> 5;
  const int lane = tid & 31;
  const int h    = lane >> 4;    // lane half (WMMA K-half select)
  const int ln   = lane & 15;    // row-within-tile (A) / column (B)
  const int mt   = wave & 3;     // which 16-row M tile
  const int np   = wave >> 2;    // which super-chunk parity

  // per-lane running top-4 (smallest keys) for each of 8 M rows
  float    bk0[8], bk1[8], bk2[8], bk3[8];
  unsigned bi0[8], bi1[8], bi2[8], bi3[8];
#pragma unroll
  for (int p = 0; p < 8; ++p) {
    bk0[p] = bk1[p] = bk2[p] = bk3[p] = 3.4e38f;
    bi0[p] = bi1[p] = bi2[p] = bi3[p] = 0u;
  }

  // cooperative B slab load: 2 phases x 32n x 32k fp32 -> f16.
  // Each thread owns 4 k-adjacent pairs (same n): 8 clause-grouped global
  // loads, then 4 packed ds_store_b32 into the [n][k] slab.
  auto loadB = [&](int si, int ks, int bufsel) {
    float ta[4], tb[4];
#pragma unroll
    for (int q = 0; q < 4; ++q) {
      int pe  = q * 256 + tid;   // pair index 0..1023
      int enp = pe >> 9;         // which phase's slab
      int rem = pe & 511;
      int ek2 = rem >> 5;        // 0..15 -> k = 2*ek2
      int en  = rem & 31;        // consecutive across lanes -> coalesced
      const float* gp = ref + (size_t)(ks * 32 + 2 * ek2) * NREF
                            + (si * 2 + enp) * SCH + en;
      ta[q] = gp[0];
      tb[q] = gp[NREF];
    }
#pragma unroll
    for (int q = 0; q < 4; ++q) {
      int pe  = q * 256 + tid;
      int enp = pe >> 9;
      int rem = pe & 511;
      int ek2 = rem >> 5;
      int en  = rem & 31;
      v2h pk = { (_Float16)ta[q], (_Float16)tb[q] };
      *(v2h*)&Bs[(enp * 2 + bufsel) * B_SLABH + en * B_STRIDE + 2 * ek2] = pk;
    }
  };

  // top-4 insert helper
  auto ins = [&](int p, float key, unsigned ng) {
    if (key < bk3[p]) {
      if (key < bk2[p]) {
        bk3[p] = bk2[p]; bi3[p] = bi2[p];
        if (key < bk1[p]) {
          bk2[p] = bk1[p]; bi2[p] = bi1[p];
          if (key < bk0[p]) { bk1[p] = bk0[p]; bi1[p] = bi0[p];
                              bk0[p] = key;    bi0[p] = ng; }
          else              { bk1[p] = key;    bi1[p] = ng; }
        } else { bk2[p] = key; bi2[p] = ng; }
      } else   { bk3[p] = key; bi3[p] = ng; }
    }
  };

  loadB(0, 0, 0);  // prologue fills buffer 0

#pragma unroll 1
  for (int i = 0; i < SCPP; ++i) {
    v8f acc0 = {}, acc1 = {};
    const int sc = i * 2 + np;           // this wave's super-chunk
    float r20 = r2[sc * SCH + ln];
    float r21 = r2[sc * SCH + NCH + ln];

    for (int k = 0; k < KSTEPS; ++k) {
      __syncthreads();                         // slab(rb) ready, readers of wb done
      const int rb = k & 1;                    // 24 toggles/chunk -> read buffer = k&1
      int ni = (k == KSTEPS - 1) ? i + 1 : i;  // prefetch next slab
      int nk = (k == KSTEPS - 1) ? 0 : k + 1;
      if (ni < SCPP) loadB(ni, nk, rb ^ 1);

      // A (16x32 f16): lanes 0-15 hold K=kb..kb+7 / kb+16..kb+23,
      // lanes 16-31 shifted by 8. Two 16B ds loads, shared by both WMMAs.
      const _Float16* ap = As + (mt * 16 + ln) * A_STRIDE + k * 32 + h * 8;
      v8h alo = *(const v8h*)ap;
      v8h ahi = *(const v8h*)(ap + 16);
      v16h av = __builtin_shufflevector(alo, ahi,
                  0,1,2,3,4,5,6,7,8,9,10,11,12,13,14,15);

      // B (32x16 f16): lane column = ln (+16 for acc1), K same order as A.
      const _Float16* bbase = Bs + (np * 2 + rb) * B_SLABH + h * 8;
      const _Float16* bp0 = bbase + ln * B_STRIDE;
      const _Float16* bp1 = bbase + (NCH + ln) * B_STRIDE;
      v8h b0lo = *(const v8h*)bp0;
      v8h b0hi = *(const v8h*)(bp0 + 16);
      v16h bv0 = __builtin_shufflevector(b0lo, b0hi,
                   0,1,2,3,4,5,6,7,8,9,10,11,12,13,14,15);
      v8h b1lo = *(const v8h*)bp1;
      v8h b1hi = *(const v8h*)(bp1 + 16);
      v16h bv1 = __builtin_shufflevector(b1lo, b1hi,
                   0,1,2,3,4,5,6,7,8,9,10,11,12,13,14,15);

      acc0 = __builtin_amdgcn_wmma_f32_16x16x32_f16(
          false, av, false, bv0, (short)0, acc0, false, false);
      acc1 = __builtin_amdgcn_wmma_f32_16x16x32_f16(
          false, av, false, bv1, (short)0, acc1, false, false);
    }

    // super-chunk complete: acc{0,1}[p] = cross(m = p + h*8, n)
    unsigned ng0 = (unsigned)(sc * SCH + ln);
    unsigned ng1 = (unsigned)(sc * SCH + NCH + ln);
#pragma unroll
    for (int p = 0; p < 8; ++p) {
      ins(p, r20 - 2.0f * acc0[p], ng0);   // ordering-equivalent to d2
      ins(p, r21 - 2.0f * acc1[p], ng1);
    }
  }

  // ---- cross-lane / cross-phase merge through LDS (reuse A region) ----
  __syncthreads();
  float*    mk = (float*)lds;                       // [64 rows][2 np][16 ln][4]
  unsigned* mi = (unsigned*)(mk + MBLK * NPH * 16 * 4);
#pragma unroll
  for (int p = 0; p < 8; ++p) {
    int m = p + h * 8;
    int o = (((mt * 16 + m) * NPH + np) * 16 + ln) * 4;
    mk[o + 0] = bk0[p]; mk[o + 1] = bk1[p]; mk[o + 2] = bk2[p]; mk[o + 3] = bk3[p];
    mi[o + 0] = bi0[p]; mi[o + 1] = bi1[p]; mi[o + 2] = bi2[p]; mi[o + 3] = bi3[p];
  }
  __syncthreads();
  if (tid < MBLK) {
    int ro = tid * (NPH * 16 * 4);
    float k0 = 3.4e38f, k1 = k0, k2 = k0, k3 = k0;
    unsigned i0 = 0, i1 = 0, i2 = 0, i3 = 0;
    for (int c = 0; c < NPH * 16 * 4; ++c) {
      float ky = mk[ro + c]; unsigned id = mi[ro + c];
      if (ky < k3) {
        if (ky < k2) { k3 = k2; i3 = i2;
          if (ky < k1) { k2 = k1; i2 = i1;
            if (ky < k0) { k1 = k0; i1 = i0; k0 = ky; i0 = id; }
            else         { k1 = ky; i1 = id; }
          } else { k2 = ky; i2 = id; }
        } else   { k3 = ky; i3 = id; }
      }
    }
    int t = wgM + tid;
    oidx[t * 4 + 0] = (int)i0; oidx[t * 4 + 1] = (int)i1;
    oidx[t * 4 + 2] = (int)i2; oidx[t * 4 + 3] = (int)i3;
  }
}

// out[d][t] = 0.25 * sum_j ref[d][idx[t][j]] — exact fp32, coalesced stores.
__global__ void gather_kernel(const float* __restrict__ ref,
                              const int* __restrict__ oidx,
                              float* __restrict__ out) {
  int t = blockIdx.x * blockDim.x + threadIdx.x;
  int d = blockIdx.y;
  int4 nn = ((const int4*)oidx)[t];
  const float* rp = ref + (size_t)d * NREF;
  out[(size_t)d * TSRC + t] =
      0.25f * (rp[nn.x] + rp[nn.y] + rp[nn.z] + rp[nn.w]);
}

extern "C" void kernel_launch(void* const* d_in, const int* in_sizes, int n_in,
                              void* d_out, int out_size, void* d_ws, size_t ws_size,
                              hipStream_t stream) {
  (void)in_sizes; (void)n_in; (void)out_size; (void)ws_size;
  const float* x   = (const float*)d_in[0];   // [768, 2048]
  const float* ref = (const float*)d_in[1];   // [768, 16384]
  float* out = (float*)d_out;

  float* r2   = (float*)d_ws;                               // 64 KB
  int*   oidx = (int*)((char*)d_ws + NREF * sizeof(float)); // 32 KB

  r2_kernel<<<NREF / 256, 256, 0, stream>>>(ref, r2);

  hipFuncSetAttribute((const void*)match_topk_kernel,
                      hipFuncAttributeMaxDynamicSharedMemorySize, LDS_BYTES);
  match_topk_kernel<<<TSRC / MBLK, 256, LDS_BYTES, stream>>>(x, ref, r2, oidx);

  gather_kernel<<<dim3(TSRC / 256, DIM), 256, 0, stream>>>(ref, oidx, out);
}
